// CustomQuantLinear_91001767068029
// MI455X (gfx1250) — compile-verified, matrix operand order checked by reading, not run on hardware
//
#include <hip/hip_runtime.h>

typedef __attribute__((ext_vector_type(16))) _Float16 v16h;
typedef __attribute__((ext_vector_type(8)))  _Float16 v8h;
typedef __attribute__((ext_vector_type(8)))  float    v8f;
typedef __attribute__((ext_vector_type(4)))  float    f32x4;
typedef __attribute__((ext_vector_type(4)))  int      i32x4;

#define OUTF 11008
#define INF  4096
#define BLKQ 128          // quant positional period (k % 128)
#define MROW 64
#define BN   32           // output channels per block
#define KC   128          // k-chunk per LDS stage (== BLKQ: thread->slot map constant)
#define LDH  (KC + 8)     // padded row stride in halfs (272 B, 16B-aligned, kills bank conflicts)

__global__ __launch_bounds__(256)
void q4_wmma_kernel(const float* __restrict__ x,
                    const int*   __restrict__ wpack,   // [OUTF][INF/2] 2 nibbles per int32
                    const float* __restrict__ scale,   // [OUTF][BLKQ]
                    const float* __restrict__ zp,      // [OUTF][BLKQ]
                    const float* __restrict__ bias,    // [OUTF]
                    float*       __restrict__ out)     // [MROW][OUTF]
{
    __shared__ _Float16 sX[MROW * LDH];   // x chunk as f16           (17.0 KB)
    __shared__ _Float16 sW[BN   * LDH];   // dequantized weights f16  ( 8.5 KB)

    const int t    = threadIdx.x;       // 0..255
    const int lane = t & 31;
    const int wv   = t >> 5;            // 0..7
    const int n0   = blockIdx.x * BN;

    // ---- prologue: fold (scale, -zp*scale) into REGISTERS for this thread's
    //      fixed dequant slots. Slot i: fid = i*256+t -> ch = fid>>4, kloc = (fid&15)*8.
    //      The (ch,kloc) map is identical for every K-chunk since KC == BLKQ.
    v8h sreg[2], breg[2];
    #pragma unroll
    for (int i = 0; i < 2; ++i) {
        int fid  = i * 256 + t;
        int ch   = fid >> 4;
        int kloc = (fid & 15) * 8;
        const float* sp = scale + (size_t)(n0 + ch) * BLKQ + kloc;
        const float* zpp = zp   + (size_t)(n0 + ch) * BLKQ + kloc;
        f32x4 s0 = *(const f32x4*)(sp);
        f32x4 s1 = *(const f32x4*)(sp + 4);
        f32x4 z0 = *(const f32x4*)(zpp);
        f32x4 z1 = *(const f32x4*)(zpp + 4);
        #pragma unroll
        for (int q = 0; q < 4; ++q) {
            sreg[i][q]     = (_Float16)s0[q];
            sreg[i][q + 4] = (_Float16)s1[q];
            breg[i][q]     = (_Float16)(-z0[q] * s0[q]);
            breg[i][q + 4] = (_Float16)(-z1[q] * s1[q]);
        }
    }

    const int m_t = (wv & 3) * 16;      // wave's M tile
    const int n_t = (wv >> 2) * 16;     // wave's N tile
    const int kb  = (lane < 16) ? 0 : 8;
    const _Float16* xa = &sX[(m_t + (lane & 15)) * LDH];
    const _Float16* wb = &sW[(n_t + (lane & 15)) * LDH];

    v8f acc = {};

    for (int kc = 0; kc < INF / KC; ++kc) {
        const int k0 = kc * KC;

        // prefetch next chunk's packed weights (global_prefetch_b8)
        if (kc + 1 < INF / KC) {
            __builtin_prefetch(wpack + (size_t)(n0 + (t >> 3)) * (INF / 2)
                                     + ((k0 + KC) >> 1) + (t & 7) * 8, 0, 0);
        }

        // ---- stage x chunk: 64x128 f32 -> f16 LDS.
        //      8 floats per slot: 2x global_load_b128 -> 1x ds_store_b128.
        #pragma unroll
        for (int i = 0; i < 4; ++i) {
            int fid = i * 256 + t;      // 0..1023
            int row = fid >> 4;         // 16 8-float groups per row
            int c8  = (fid & 15) * 8;
            const float* xp = x + (size_t)row * INF + k0 + c8;
            f32x4 a = *(const f32x4*)(xp);
            f32x4 b = *(const f32x4*)(xp + 4);
            v8h h;
            #pragma unroll
            for (int q = 0; q < 4; ++q) {
                h[q]     = (_Float16)a[q];
                h[q + 4] = (_Float16)b[q];
            }
            *(v8h*)(&sX[row * LDH + c8]) = h;   // 16B store
        }

        // ---- stage weights: 32ch x 64 words; dequant with register f16 scales ----
        #pragma unroll
        for (int i = 0; i < 2; ++i) {
            int fid  = i * 256 + t;     // 0..511
            int ch   = fid >> 4;        // 16 int4 per channel
            int kloc = (fid & 15) * 8;  // 4 words -> 8 unpacked k values
            i32x4 w = *(const i32x4*)(wpack + (size_t)(n0 + ch) * (INF / 2)
                                            + (k0 >> 1) + (kloc >> 1));
            v8h nv;
            #pragma unroll
            for (int q = 0; q < 4; ++q) {
                int word = w[q];
                nv[q * 2 + 0] = (_Float16)((word >> 4) & 15);  // high nibble (k even)
                nv[q * 2 + 1] = (_Float16)(word & 15);         // low nibble  (k odd)
            }
            v8h hv = nv * sreg[i] + breg[i];    // v_pk_fma_f16 x4
            *(v8h*)(&sW[ch * LDH + kloc]) = hv; // 16B store
        }
        __syncthreads();

        // ---- 4 WMMAs per wave over the 128-wide chunk ----
        #pragma unroll
        for (int kk = 0; kk < KC; kk += 32) {
            v8h alo = *(const v8h*)(xa + kk + kb);
            v8h ahi = *(const v8h*)(xa + kk + kb + 16);
            v8h blo = *(const v8h*)(wb + kk + kb);
            v8h bhi = *(const v8h*)(wb + kk + kb + 16);
            v16h a, b;
            #pragma unroll
            for (int q = 0; q < 8; ++q) {
                a[q] = alo[q]; a[q + 8] = ahi[q];
                b[q] = blo[q]; b[q + 8] = bhi[q];
            }
            acc = __builtin_amdgcn_wmma_f32_16x16x32_f16(
                      false, a, false, b, (short)0, acc, false, false);
        }
        __syncthreads();
    }

    // ---- writeback: D layout = VGPR r -> M=(lane<16?0:8)+r, N=lane&15 ----
    const int col   = n0 + n_t + (lane & 15);
    const float bv  = bias[col];
    const int rbase = m_t + ((lane < 16) ? 0 : 8);
    #pragma unroll
    for (int r = 0; r < 8; ++r) {
        out[(size_t)(rbase + r) * OUTF + col] = acc[r] + bv;
    }
}

extern "C" void kernel_launch(void* const* d_in, const int* in_sizes, int n_in,
                              void* d_out, int out_size, void* d_ws, size_t ws_size,
                              hipStream_t stream) {
    const float* x     = (const float*)d_in[0];
    const int*   wpk   = (const int*)d_in[1];
    const float* scale = (const float*)d_in[2];
    const float* zp    = (const float*)d_in[3];
    const float* bias  = (const float*)d_in[4];
    float* out = (float*)d_out;

    dim3 grid(OUTF / BN);   // 344 blocks
    dim3 block(256);        // 8 wave32
    q4_wmma_kernel<<<grid, block, 0, stream>>>(x, wpk, scale, zp, bias, out);
}
